// GraphTransformerEncoder_3418793967881
// MI455X (gfx1250) — matrix operand order repaired and checked by reading, not compiled, not run on hardware
//
#include <hip/hip_runtime.h>
#include <math.h>

// ---------------- CDNA5 WMMA types ----------------
typedef __attribute__((ext_vector_type(16))) __bf16 v16bf;
typedef __attribute__((ext_vector_type(8)))  float  v8f;
typedef unsigned short ushort_t;

static constexpr int NN = 50000;   // nodes
static constexpr int NE = 800000;  // edges
static constexpr int CH = 128;     // channels (both layers)

__device__ __forceinline__ ushort_t f32_to_bf16(float f) {
    unsigned u = __float_as_uint(f);
    u += 0x7FFFu + ((u >> 16) & 1u);   // round to nearest even
    return (ushort_t)(u >> 16);
}

__device__ __forceinline__ void atomicMaxFloat(float* addr, float val) {
    // valid when *addr initialized to -inf; IEEE monotonic int trick
    if (val >= 0.f) atomicMax((int*)addr, __float_as_int(val));
    else            atomicMin((unsigned int*)addr, __float_as_uint(val));
}

// -------- pack 128x128 f32 weight into bf16 WMMA-B fragment order --------
// Bp[((kt*8+ntile)*32+lane)*16 + j] = bf16(W[k][n])
//   n = ntile*16 + (lane&15);  k = kt*32 + (lane&16 ? 16:0) + j   (contiguous K order, same as the A loads)
__global__ void pack_weight_kernel(const float* __restrict__ W, ushort_t* __restrict__ Wp) {
    int t = blockIdx.x * blockDim.x + threadIdx.x;   // 0..16383
    int j     =  t        & 15;
    int lane  = (t >> 4)  & 31;
    int ntile = (t >> 9)  & 7;
    int kt    =  t >> 12;
    int n = ntile * 16 + (lane & 15);
    int k = kt * 32 + ((lane & 16) ? 16 : 0) + j;
    Wp[t] = f32_to_bf16(W[k * CH + n]);
}

__global__ void cvt_bf16_kernel(const float* __restrict__ in, ushort_t* __restrict__ out, int n) {
    int t = blockIdx.x * blockDim.x + threadIdx.x;
    if (t < n) out[t] = f32_to_bf16(in[t]);
}

// -------- fused 4-projection GEMM:  {Q,K,V,S}[N x128] = A_bf16 @ Wpack[w] + bias[w] ----
// block = 256 (8 waves). One 16-row A tile staged in LDS once per block; each wave
// owns one of the 8 output column tiles and runs 4 weights x 4 K-tiles = 16 WMMAs.
__global__ void gemm_qkvs_kernel(const ushort_t* __restrict__ A,
                                 const ushort_t* __restrict__ Bp,   // 4 packed weights, contiguous
                                 const float* __restrict__ b0, const float* __restrict__ b1,
                                 const float* __restrict__ b2, const float* __restrict__ b3,
                                 float* __restrict__ o0, float* __restrict__ o1,
                                 float* __restrict__ o2, float* __restrict__ o3) {
    __shared__ ushort_t sA[16 * CH];                  // 4KB bf16 tile
    int row0 = blockIdx.x * 16;
    {   // cooperative stage: 256 threads x one uint4 (16B) = 4KB
        int t = threadIdx.x;
        int r = t >> 4, chunk = t & 15;               // 16 rows x 16 chunks
        ((uint4*)sA)[t] = *(const uint4*)(A + (size_t)(row0 + r) * CH + chunk * 8);
    }
    __syncthreads();

    int wave = threadIdx.x >> 5;                      // 0..7 -> output column tile
    int lane = threadIdx.x & 31;
    int m    = lane & 15;
    int koff = (lane & 16) ? 16 : 0;                  // contiguous K order, matches packed B

    union Frag { v16bf v; uint4 q[2]; };
    Frag a[4];
#pragma unroll
    for (int kt = 0; kt < 4; ++kt) {
        int k0 = kt * 32;
        a[kt].q[0] = *(const uint4*)(sA + m * CH + k0 + koff);        // K = k0+koff..+7
        a[kt].q[1] = *(const uint4*)(sA + m * CH + k0 + 8 + koff);    // K = k0+8+koff..+7
    }

    const float* biases[4] = {b0, b1, b2, b3};
    float*       outs[4]   = {o0, o1, o2, o3};
    int col   = wave * 16 + m;
    int rbase = row0 + ((lane & 16) ? 8 : 0);         // ISA f32 C/D layout

#pragma unroll
    for (int w = 0; w < 4; ++w) {
        const ushort_t* Bw = Bp + (size_t)w * 16384;
        v8f c = {0.f,0.f,0.f,0.f,0.f,0.f,0.f,0.f};
#pragma unroll
        for (int kt = 0; kt < 4; ++kt) {
            Frag b;
            const ushort_t* bp = Bw + (((kt * 8 + wave) * 32 + lane) * 16);
            b.q[0] = *(const uint4*)(bp);
            b.q[1] = *(const uint4*)(bp + 8);
            c = __builtin_amdgcn_wmma_f32_16x16x32_bf16(false, a[kt].v, false, b.v,
                                                        (short)0, c, false, false);
        }
        float bi = biases[w][col];
        float* C = outs[w];
#pragma unroll
        for (int r = 0; r < 8; ++r)
            C[(size_t)(rbase + r) * CH + col] = c[r] + bi;
    }
}

// -------- reset per-layer stats --------
__global__ void init_stats_kernel(float* __restrict__ agg, float* __restrict__ mx,
                                  float* __restrict__ denom, int n128, int nH) {
    int t = blockIdx.x * blockDim.x + threadIdx.x;
    if (t < n128) agg[t] = 0.f;
    if (t < nH) { mx[t] = -INFINITY; denom[t] = 0.f; }
}

// -------- pass 1: alpha[e,h] = q[dst]·(k[src]+e)/sqrt(C); atomic max per (dst,h) ----
template<int H, int C>
__global__ void edge_alpha_kernel(const long long* __restrict__ srcI,
                                  const long long* __restrict__ dstI,
                                  const float* __restrict__ ea,
                                  const float* __restrict__ We,   // [16][H*C]
                                  const float* __restrict__ q,
                                  const float* __restrict__ k,
                                  float* __restrict__ alpha,
                                  float* __restrict__ mx,
                                  float scale) {
    __shared__ float sWe[16 * 128];
    for (int i = threadIdx.x; i < 16 * 128; i += blockDim.x) sWe[i] = We[i];
    __syncthreads();
    int t = blockIdx.x * blockDim.x + threadIdx.x;
    if (t >= NE * H) return;
    int h = t % H;
    int e = t / H;
    int s = (int)srcI[e], d = (int)dstI[e];
    float av[16];
#pragma unroll
    for (int dd = 0; dd < 16; ++dd) av[dd] = ea[(size_t)e * 16 + dd];
    const float* qrow = q + (size_t)d * CH + h * C;
    const float* krow = k + (size_t)s * CH + h * C;
    float acc = 0.f;
    for (int c = 0; c < C; ++c) {
        float ec = 0.f;
#pragma unroll
        for (int dd = 0; dd < 16; ++dd) ec += av[dd] * sWe[dd * CH + h * C + c];
        acc += qrow[c] * (krow[c] + ec);
    }
    acc *= scale;
    alpha[t] = acc;
    atomicMaxFloat(&mx[(size_t)d * H + h], acc);
}

// -------- pass 2: p = exp(alpha-m); denom += p; agg[dst] += p*(v[src]+e) (unnormalized)
template<int H, int C>
__global__ void edge_msg_kernel(const long long* __restrict__ srcI,
                                const long long* __restrict__ dstI,
                                const float* __restrict__ ea,
                                const float* __restrict__ We,
                                const float* __restrict__ v,
                                const float* __restrict__ alpha,
                                const float* __restrict__ mx,
                                float* __restrict__ denom,
                                float* __restrict__ agg) {
    __shared__ float sWe[16 * 128];
    for (int i = threadIdx.x; i < 16 * 128; i += blockDim.x) sWe[i] = We[i];
    __syncthreads();
    int t = blockIdx.x * blockDim.x + threadIdx.x;
    if (t >= NE * H) return;
    int h = t % H;
    int e = t / H;
    int s = (int)srcI[e], d = (int)dstI[e];
    float p = __expf(alpha[t] - mx[(size_t)d * H + h]);
    atomicAdd(&denom[(size_t)d * H + h], p);
    float av[16];
#pragma unroll
    for (int dd = 0; dd < 16; ++dd) av[dd] = ea[(size_t)e * 16 + dd];
    const float* vrow = v + (size_t)s * CH + h * C;
    float* arow = agg + (size_t)d * CH + h * C;
    for (int c = 0; c < C; ++c) {
        float ec = 0.f;
#pragma unroll
        for (int dd = 0; dd < 16; ++dd) ec += av[dd] * sWe[dd * CH + h * C + c];
        atomicAdd(&arow[c], p * (vrow[c] + ec));
    }
}

// -------- finalize layer0: h = relu(agg/denom + skip) -> bf16 for next layer ------
__global__ void finalize0_kernel(const float* __restrict__ agg, const float* __restrict__ denom,
                                 const float* __restrict__ s, ushort_t* __restrict__ xb) {
    int t = blockIdx.x * blockDim.x + threadIdx.x;   // NN*128
    int n = t >> 7, hc = t & 127, h = hc >> 4;
    float val = agg[t] / (denom[n * 8 + h] + 1e-16f) + s[t];
    xb[t] = f32_to_bf16(fmaxf(val, 0.f));
}

// -------- finalize layer1: out = agg/denom + skip (H=1) ------
__global__ void finalize1_kernel(const float* __restrict__ agg, const float* __restrict__ denom,
                                 const float* __restrict__ s, float* __restrict__ out) {
    int t = blockIdx.x * blockDim.x + threadIdx.x;   // NN*128
    int n = t >> 7;
    out[t] = agg[t] / (denom[n] + 1e-16f) + s[t];
}

extern "C" void kernel_launch(void* const* d_in, const int* in_sizes, int n_in,
                              void* d_out, int out_size, void* d_ws, size_t ws_size,
                              hipStream_t stream) {
    const float*     x    = (const float*)d_in[0];
    const long long* eidx = (const long long*)d_in[1];      // int64 [2][E]
    const float*     ea   = (const float*)d_in[2];
    const float* Wq0 = (const float*)d_in[3];  const float* bq0 = (const float*)d_in[4];
    const float* Wk0 = (const float*)d_in[5];  const float* bk0 = (const float*)d_in[6];
    const float* Wv0 = (const float*)d_in[7];  const float* bv0 = (const float*)d_in[8];
    const float* We0 = (const float*)d_in[9];
    const float* Ws0 = (const float*)d_in[10]; const float* bs0 = (const float*)d_in[11];
    const float* Wq1 = (const float*)d_in[12]; const float* bq1 = (const float*)d_in[13];
    const float* Wk1 = (const float*)d_in[14]; const float* bk1 = (const float*)d_in[15];
    const float* Wv1 = (const float*)d_in[16]; const float* bv1 = (const float*)d_in[17];
    const float* We1 = (const float*)d_in[18];
    const float* Ws1 = (const float*)d_in[19]; const float* bs1 = (const float*)d_in[20];

    const long long* srcI = eidx;
    const long long* dstI = eidx + NE;

    // ---- workspace carve-out ----
    char* ws = (char*)d_ws;
    size_t off = 0;
    auto carve = [&](size_t bytes) { char* p = ws + off; off = (off + bytes + 255) & ~(size_t)255; return p; };
    ushort_t* wpack = (ushort_t*)carve(8 * 16384 * sizeof(ushort_t)); // 8 packed 128x128 bf16
    ushort_t* xb    = (ushort_t*)carve((size_t)NN * CH * sizeof(ushort_t));
    float* q_  = (float*)carve((size_t)NN * CH * 4);
    float* k_  = (float*)carve((size_t)NN * CH * 4);
    float* v_  = (float*)carve((size_t)NN * CH * 4);
    float* s_  = (float*)carve((size_t)NN * CH * 4);
    float* mx  = (float*)carve((size_t)NN * 8 * 4);
    float* dn  = (float*)carve((size_t)NN * 8 * 4);
    float* alp = (float*)carve((size_t)NE * 8 * 4);
    float* agg = (float*)carve((size_t)NN * CH * 4);
    (void)ws_size; (void)in_sizes; (void)n_in; (void)out_size;

    const int B = 256;
    const int gNode  = (NN * CH) / B;   // 25000
    const int gTiles = NN / 16;         // 3125
    const int gE8    = (NE * 8) / B;    // 25000
    const int gE1    = (NE * 1) / B;    // 3125

    // ---- pack all 8 node-weight matrices into WMMA fragment order (bf16) ----
    const float* Wn[8] = {Wq0, Wk0, Wv0, Ws0, Wq1, Wk1, Wv1, Ws1};
    for (int i = 0; i < 8; ++i)
        pack_weight_kernel<<<64, B, 0, stream>>>(Wn[i], wpack + (size_t)i * 16384);

    // ================= Layer 0 (H=8, C=16) =================
    cvt_bf16_kernel<<<gNode, B, 0, stream>>>(x, xb, NN * CH);
    gemm_qkvs_kernel<<<gTiles, B, 0, stream>>>(xb, wpack,
                                               bq0, bk0, bv0, bs0, q_, k_, v_, s_);
    init_stats_kernel<<<gNode, B, 0, stream>>>(agg, mx, dn, NN * CH, NN * 8);
    edge_alpha_kernel<8, 16><<<gE8, B, 0, stream>>>(srcI, dstI, ea, We0, q_, k_, alp, mx, 0.25f);
    edge_msg_kernel<8, 16><<<gE8, B, 0, stream>>>(srcI, dstI, ea, We0, v_, alp, mx, dn, agg);
    finalize0_kernel<<<gNode, B, 0, stream>>>(agg, dn, s_, xb);   // xb <- bf16(relu(h))

    // ================= Layer 1 (H=1, C=128) =================
    gemm_qkvs_kernel<<<gTiles, B, 0, stream>>>(xb, wpack + (size_t)4 * 16384,
                                               bq1, bk1, bv1, bs1, q_, k_, v_, s_);
    init_stats_kernel<<<gNode, B, 0, stream>>>(agg, mx, dn, NN * CH, NN * 1);
    edge_alpha_kernel<1, 128><<<gE1, B, 0, stream>>>(srcI, dstI, ea, We1, q_, k_, alp, mx,
                                                     0.088388347648318447f /* 1/sqrt(128) */);
    edge_msg_kernel<1, 128><<<gE1, B, 0, stream>>>(srcI, dstI, ea, We1, v_, alp, mx, dn, agg);
    finalize1_kernel<<<gNode, B, 0, stream>>>(agg, dn, s_, (float*)d_out);
}
